// SMModel_55585466744992
// MI455X (gfx1250) — compile-verified, matrix operand order checked by reading, not run on hardware
//
#include <hip/hip_runtime.h>
#include <hip/hip_bf16.h>
#include <math.h>

// ---------------------------------------------------------------------------
// SMModel for MI455X (gfx1250, wave32, WMMA).
//
// conv1(3x3) -> conv2(3x3 grouped), linear in between, collapses into one
// 5x5 conv producing the predicted per-pixel kernels Kf directly:
//   Weff[q][5][5] = sum_hid W2[q,hid] (*) W1[group(q),hid]
//   beff[q]       = b2[q] + sum_{hid,a,b} W2[q,hid,a,b]*b1[group(q)*HID+hid]
// (~8x fewer FLOPs, no (B,1280,N,N) intermediate).  The 5x5 conv is a GEMM
//   Kf[q,pix] = Weff[q,0..24] * patch[0..24,pix]    (+ bias via K column 25)
// run on V_WMMA_F32_16X16X32_BF16 (K = 25 + 1 bias + 6 zero = 32, bf16 in /
// f32 accumulate).  Kf rows inside each output channel o are permuted as
// row = k9*NCIN + i  (k9 = 3x3 tap, i = input channel), padded to 16*TPO.
// For NCIN=8 a consumed row sub*16 + hi*8 + r has i = r (compile-time) and
// k9 = 2*sub + hi (uniform per lane half), so the per-pixel einsum is
// consumed branch-free straight out of the WMMA accumulator VGPRs:
// one cndmask per sub + 8 imm-offset ds_load_b32 + 8 v_fmac.
// ---------------------------------------------------------------------------

#define NC 8
#define HID 20
#define KS2 9

typedef __attribute__((ext_vector_type(16))) __bf16 v16bf;
typedef __attribute__((ext_vector_type(8)))  float  v8f;

union Frag {
  v16bf v;
  unsigned short s[16];
  uint4 q[2];
};

static __device__ inline unsigned short f2bf(float f) {
  unsigned int u = __float_as_uint(f);
  unsigned int r = u + 0x7FFFu + ((u >> 16) & 1u);   // round-to-nearest-even
  return (unsigned short)(r >> 16);
}

static __device__ inline float elu1(float v) {
  return v > 0.f ? v : (__expf(v) - 1.f);
}

// ---------------------------------------------------------------------------
// Compose effective 5x5 conv + bias; scatter into WMMA A-fragment (16x32 bf16)
// lane layout.  Row layout inside each output channel o: row = k9*nc_in + i,
// rows with k9 == 9 (i.e. row >= 9*nc_in) are zero padding.
//   lo lanes (0-15):  vgpr0..3 -> K 0..7,  vgpr4..7 -> K 16..23
//   hi lanes (16-31): vgpr0..3 -> K 8..15, vgpr4..7 -> K 24..31
// One thread per (tile, row m, K col k).  K col 25 = bias, 26..31 = zero.
// ---------------------------------------------------------------------------
__global__ __launch_bounds__(256) void compose_kernel(
    const float* __restrict__ W1, const float* __restrict__ b1,
    const float* __restrict__ W2, const float* __restrict__ b2,
    unsigned short* __restrict__ Apack, int nc_in, int TPO)
{
  int gid = blockIdx.x * 256 + threadIdx.x;
  int tiles = NC * TPO;
  if (gid >= tiles * 512) return;
  int tile = gid >> 9;
  int rem  = gid & 511;
  int m = rem >> 5;          // row within 16-row tile
  int k = rem & 31;          // K column 0..31
  int o = tile / TPO;
  int sub = tile - o * TPO;
  int row = sub * 16 + m;    // row within this output channel: k9*nc_in + i
  int k9  = row / nc_in;
  int i   = row - k9 * nc_in;

  float val = 0.f;
  if (k9 < 9 && k <= 25) {
    int g  = o * nc_in + i;        // conv2 group
    int q2 = g * 9 + k9;           // row of W2 / b2
    int c1 = g * HID;              // base channel of W1 / b1
    if (k == 25) {                 // effective bias
      float s = b2[q2];
      for (int hid = 0; hid < HID; ++hid) {
        const float* w2p = W2 + (q2 * HID + hid) * 9;
        float ws = 0.f;
        for (int t9 = 0; t9 < 9; ++t9) ws += w2p[t9];
        s += ws * b1[c1 + hid];
      }
      val = s;
    } else {                       // effective 5x5 weight
      int u = k / 5, v = k - u * 5;
      float s = 0.f;
      for (int a = 0; a < 3; ++a) {
        int c = u - a; if (c < 0 || c > 2) continue;
        for (int bb = 0; bb < 3; ++bb) {
          int d = v - bb; if (d < 0 || d > 2) continue;
          const float* w2p = W2 + (q2 * HID) * 9 + a * 3 + bb;
          const float* w1p = W1 + c1 * 9 + c * 3 + d;
          float acc = 0.f;
          for (int hid = 0; hid < HID; ++hid)
            acc += w2p[hid * 9] * w1p[hid * 9];
          s += acc;
        }
      }
      val = s;
    }
  }
  int half = (k >> 3) & 1;
  int lane = m + half * 16;
  int vg   = ((k & 7) >> 1) + ((k & 16) ? 4 : 0);
  int slot = vg * 2 + (k & 1);
  Apack[(tile * 32 + lane) * 16 + slot] = f2bf(val);
}

// ---------------------------------------------------------------------------
// Fused sm_block.  Grid: (N/16, N/8, B), 256 threads = 8 waves; wave w owns
// row y0+w, 16 pixels wide.  Per wave: one B fragment (5x5 patches + bias 1),
// then for each o: TPO WMMAs against prepacked A tiles; C fragments consumed
// branch-free (NCIN=8) against LDS-staged x patches.
// ---------------------------------------------------------------------------
template<int NCIN, int TPO>
__global__ __launch_bounds__(256) void sm_apply_kernel(
    const float* __restrict__ img,                 // (8, N, N)
    const float* __restrict__ xin,                 // (8, NCIN, N, N)
    const unsigned short* __restrict__ Apack,      // NC*TPO tiles * 512 bf16
    float* __restrict__ out,                       // (8, NC, N, N)
    int N)
{
  __shared__ float imgT[12][20];        // 16x8 tile + 2-halo (5x5 patches)
  __shared__ float xT[NCIN][10][18];    // 16x8 tile + 1-halo (3x3 patches)

  int tid = threadIdx.x;
  int b  = blockIdx.z;
  int x0 = blockIdx.x * 16;
  int y0 = blockIdx.y * 8;

  const float* imgb = img + (size_t)b * N * N;
  for (int idx = tid; idx < 240; idx += 256) {
    int ry = idx / 20, rx = idx - ry * 20;
    int gy = y0 + ry - 2, gx = x0 + rx - 2;
    float v = 0.f;
    if (gy >= 0 && gy < N && gx >= 0 && gx < N) v = imgb[gy * N + gx];
    imgT[ry][rx] = v;
  }
  for (int idx = tid; idx < NCIN * 180; idx += 256) {
    int i = idx / 180; int r2 = idx - i * 180;
    int ry = r2 / 18, rx = r2 - ry * 18;
    int gy = y0 + ry - 1, gx = x0 + rx - 1;
    float v = 0.f;
    if (gy >= 0 && gy < N && gx >= 0 && gx < N)
      v = xin[(((size_t)b * NCIN + i) * N + gy) * N + gx];
    xT[i][ry][rx] = v;
  }
  __syncthreads();

  int w    = tid >> 5;        // wave id = row within tile
  int lane = tid & 31;
  int pix  = lane & 15;       // pixel column (WMMA N index)
  int hi   = lane >> 4;       // lane half selects K rows 8..15 / 24..31

  // B fragment: column pix = 5x5 image patch (K 0..24), K25 = 1.0 (bias),
  // K26..31 = 0.  Built once, reused by all NC*TPO WMMAs.
  Frag bfr;
#pragma unroll
  for (int s = 0; s < 16; ++s) {
    int vg = s >> 1, odd = s & 1;
    int k = (vg < 4 ? vg * 2 : 16 + (vg - 4) * 2) + hi * 8 + odd;
    float v;
    if (k < 25) {
      int u = k / 5, vv = k - u * 5;          // dy = u-2, dx = vv-2
      v = imgT[w + u][pix + vv];
    } else {
      v = (k == 25) ? 1.0f : 0.f;
    }
    bfr.s[s] = f2bf(v);
  }

  // LDS offset of 3x3 tap k9 relative to &xT[i][w][pix]:  g(k9)=(k9/3)*18+k9%3
  constexpr int GL[5] = {0, 2, 19, 36, 38};  // g(2*sub)
  constexpr int GH[5] = {1, 18, 20, 37, 0};  // g(2*sub+1); sub=4 -> pad (safe 0)
  constexpr int G9[9] = {0, 1, 2, 18, 19, 20, 36, 37, 38};

  const float* xbase = &xT[0][w][pix];
  const uint4* apl = (const uint4*)Apack + lane * 2;

  for (int o = 0; o < NC; ++o) {
    if (o + 1 < NC)  // prefetch next o's A fragments (global_prefetch_b8)
      __builtin_prefetch((const void*)(apl + (o + 1) * TPO * 64), 0, 3);
    float acc = 0.f;
#pragma unroll
    for (int sub = 0; sub < TPO; ++sub) {
      Frag af;
      af.q[0] = apl[(o * TPO + sub) * 64 + 0];
      af.q[1] = apl[(o * TPO + sub) * 64 + 1];
      v8f c = {};
      c = __builtin_amdgcn_wmma_f32_16x16x32_bf16(
          false, af.v, false, bfr.v, (short)0, c, false, false);
      // c[r] = Kf[row = sub*16 + hi*8 + r][pixel], row = k9*NCIN + i.
      if (NCIN == 1) {
        // init block: k9 = row directly; hi half has only row 8 (k9=8) valid.
        if (hi) {
          acc += c[0] * xbase[38];
        } else {
#pragma unroll
          for (int r = 0; r < 8; ++r) acc += c[r] * xbase[G9[r]];
        }
      } else {
        // i = r, k9 = 2*sub + hi: one select per sub, then branch-free
        // imm-offset ds_load + fmac.  Pad rows (sub=4,hi) have zero Kf.
        const float* xs = xbase + (hi ? GH[sub] : GL[sub]);
#pragma unroll
        for (int r = 0; r < 8; ++r) acc += c[r] * xs[r * 180];
      }
    }
    float tot = acc + __shfl_xor(acc, 16, 32);   // rows 0-7 + rows 8-15
    if (lane < 16) {
      out[(((size_t)b * NC + o) * N + (y0 + w)) * N + (x0 + pix)] = elu1(tot);
    }
  }
}

// 2x2 average pool, NCHW, one thread per output element.
__global__ __launch_bounds__(256) void down_kernel(
    const float* __restrict__ in, float* __restrict__ out, int C, int No)
{
  int idx = blockIdx.x * 256 + threadIdx.x;
  int total = 8 * C * No * No;
  if (idx >= total) return;
  int x = idx % No; int t = idx / No;
  int y = t % No;   t /= No;                 // t = b*C + c
  int Ni = No * 2;
  const float* p = in + ((size_t)t * Ni + 2 * y) * Ni + 2 * x;
  out[idx] = 0.25f * (p[0] + p[1] + p[Ni] + p[Ni + 1]);
}

// x + up(x1) + up(up(x2)) at full resolution (128).
__global__ __launch_bounds__(256) void combine_kernel(
    const float* __restrict__ xf, const float* __restrict__ x1,
    const float* __restrict__ x2, float* __restrict__ out)
{
  int idx = blockIdx.x * 256 + threadIdx.x;
  if (idx >= 8 * NC * 128 * 128) return;
  int x = idx & 127; int t = idx >> 7;
  int y = t & 127;   int bc = t >> 7;
  out[idx] = xf[idx]
           + x1[((size_t)bc * 64 + (y >> 1)) * 64 + (x >> 1)]
           + x2[((size_t)bc * 32 + (y >> 2)) * 32 + (x >> 2)];
}

// Final 1x1 heads: out = w6 . elu(w5 . x + b5) + b6, one thread per pixel.
__global__ __launch_bounds__(256) void head_kernel(
    const float* __restrict__ xc, const float* __restrict__ w5,
    const float* __restrict__ b5, const float* __restrict__ w6,
    const float* __restrict__ b6, float* __restrict__ out)
{
  int idx = blockIdx.x * 256 + threadIdx.x;
  if (idx >= 8 * 128 * 128) return;
  int b = idx / (128 * 128);
  int p = idx - b * 128 * 128;
  float v[NC];
#pragma unroll
  for (int c = 0; c < NC; ++c) v[c] = xc[((size_t)b * NC + c) * 128 * 128 + p];
  float r = b6[0];
#pragma unroll
  for (int o = 0; o < NC; ++o) {
    float s = b5[o];
#pragma unroll
    for (int c = 0; c < NC; ++c) s += w5[o * NC + c] * v[c];
    r += w6[o] * elu1(s);
  }
  out[idx] = r;
}

extern "C" void kernel_launch(void* const* d_in, const int* in_sizes, int n_in,
                              void* d_out, int out_size, void* d_ws, size_t ws_size,
                              hipStream_t stream) {
  const float* image   = (const float*)d_in[0];   // (8,128,128)
  const float* x_in    = (const float*)d_in[1];   // (8,1,128,128)
  const float* init_W1 = (const float*)d_in[2];
  const float* init_b1 = (const float*)d_in[3];
  const float* init_W2 = (const float*)d_in[4];
  const float* init_b2 = (const float*)d_in[5];
  const float* blk_W1  = (const float*)d_in[6];   // (12,1280,1,3,3)
  const float* blk_b1  = (const float*)d_in[7];   // (12,1280)
  const float* blk_W2  = (const float*)d_in[8];   // (12,576,20,3,3)
  const float* blk_b2  = (const float*)d_in[9];   // (12,576)
  const float* w5      = (const float*)d_in[10];
  const float* b5      = (const float*)d_in[11];
  const float* w6      = (const float*)d_in[12];
  const float* b6      = (const float*)d_in[13];
  float* out = (float*)d_out;

  // Workspace layout (all written before read; no reliance on poison state).
  char* ws = (char*)d_ws;
  size_t off = 0;
  auto alloc = [&](size_t bytes) { void* p = ws + off; off = (off + bytes + 255) & ~(size_t)255; return p; };
  unsigned short* A_init = (unsigned short*)alloc(8 * 512 * sizeof(unsigned short));        // 8 tiles
  unsigned short* A_blk  = (unsigned short*)alloc(12 * 40 * 512 * sizeof(unsigned short));  // 40 tiles each
  float* img1 = (float*)alloc(8 * 64 * 64 * sizeof(float));
  float* img2 = (float*)alloc(8 * 32 * 32 * sizeof(float));
  float* xA   = (float*)alloc((size_t)8 * NC * 128 * 128 * sizeof(float));
  float* xB   = (float*)alloc((size_t)8 * NC * 128 * 128 * sizeof(float));
  float* x1A  = (float*)alloc((size_t)8 * NC * 64 * 64 * sizeof(float));
  float* x1B  = (float*)alloc((size_t)8 * NC * 64 * 64 * sizeof(float));
  float* x2A  = (float*)alloc((size_t)8 * NC * 32 * 32 * sizeof(float));
  float* x2B  = (float*)alloc((size_t)8 * NC * 32 * 32 * sizeof(float));

  // --- Compose effective 5x5 conv weights into prepacked WMMA A fragments ---
  compose_kernel<<<(8 * 512 + 255) / 256, 256, 0, stream>>>(
      init_W1, init_b1, init_W2, init_b2, A_init, /*nc_in=*/1, /*TPO=*/1);
  for (int k = 0; k < 12; ++k) {
    compose_kernel<<<(40 * 512 + 255) / 256, 256, 0, stream>>>(
        blk_W1 + (size_t)k * 1280 * 9, blk_b1 + (size_t)k * 1280,
        blk_W2 + (size_t)k * 576 * HID * 9, blk_b2 + (size_t)k * 576,
        A_blk + (size_t)k * 40 * 512, /*nc_in=*/NC, /*TPO=*/5);
  }

  // --- Image pyramid ---
  down_kernel<<<(8 * 1 * 64 * 64 + 255) / 256, 256, 0, stream>>>(image, img1, 1, 64);
  down_kernel<<<(8 * 1 * 32 * 32 + 255) / 256, 256, 0, stream>>>(img1, img2, 1, 32);

  // --- Init sm_block (nc_in=1) + downsampled copies ---
  sm_apply_kernel<1, 1><<<dim3(128 / 16, 128 / 8, 8), 256, 0, stream>>>(
      image, x_in, A_init, xA, 128);
  down_kernel<<<(8 * NC * 64 * 64 + 255) / 256, 256, 0, stream>>>(xA, x1A, NC, 64);
  down_kernel<<<(8 * NC * 32 * 32 + 255) / 256, 256, 0, stream>>>(x1A, x2A, NC, 32);

  // --- 4 rounds of 3-scale sm_blocks (ping-pong buffers) ---
  float *xc = xA,  *xo = xB;
  float *x1c = x1A, *x1o = x1B;
  float *x2c = x2A, *x2o = x2B;
  for (int t = 0; t < 4; ++t) {
    const unsigned short* A0 = A_blk + (size_t)(3 * t + 0) * 40 * 512;
    const unsigned short* A1 = A_blk + (size_t)(3 * t + 1) * 40 * 512;
    const unsigned short* A2 = A_blk + (size_t)(3 * t + 2) * 40 * 512;
    sm_apply_kernel<NC, 5><<<dim3(128 / 16, 128 / 8, 8), 256, 0, stream>>>(image, xc,  A0, xo,  128);
    sm_apply_kernel<NC, 5><<<dim3(64 / 16,  64 / 8,  8), 256, 0, stream>>>(img1,  x1c, A1, x1o, 64);
    sm_apply_kernel<NC, 5><<<dim3(32 / 16,  32 / 8,  8), 256, 0, stream>>>(img2,  x2c, A2, x2o, 32);
    float* tswap;
    tswap = xc;  xc = xo;   xo = tswap;
    tswap = x1c; x1c = x1o; x1o = tswap;
    tswap = x2c; x2c = x2o; x2o = tswap;
  }

  // --- Multi-scale merge + heads ---
  combine_kernel<<<((8 * NC * 128 * 128) + 255) / 256, 256, 0, stream>>>(xc, x1c, x2c, xo);
  head_kernel<<<((8 * 128 * 128) + 255) / 256, 256, 0, stream>>>(xo, w5, b5, w6, b6, out);
}
